// AtomicKANLayer_59639915872404
// MI455X (gfx1250) — matrix operand order repaired and checked by reading, not compiled, not run on hardware
//
#include <hip/hip_runtime.h>
#include <hip/hip_bf16.h>

typedef __attribute__((ext_vector_type(16))) _Float16 v16h;
typedef __attribute__((ext_vector_type(8)))  float    v8f;

#define BDIM   256      // batch
#define IDIM   128      // input features
#define ODIM   256      // output features
#define DDIM   33       // degree+1 centers
#define KDIM   (IDIM * DDIM)   // 4224 = 132 * 32
#define NSUM   100
#define NPROD  10
#define A_SUP  1.5f     // support half-width a = (n+2)/2, n=1
#define INV2PI 0.15915494309189535f

union FragH { v16h v; float4 f4[2]; };

// hardware sinc: sin(u)/u via v_sin_f32 (input in revolutions)
__device__ __forceinline__ float hsinc(float u) {
    return __builtin_amdgcn_sinf(u * INV2PI) / u;
}

// ---------------------------------------------------------------------------
// Kernel 0: zero the basis workspace with wide stores (written sparsely later)
// ---------------------------------------------------------------------------
__global__ void zero_basis(float4* __restrict__ p) {
    const int idx = blockIdx.x * blockDim.x + threadIdx.x;
    float4 z; z.x = 0.f; z.y = 0.f; z.z = 0.f; z.w = 0.f;
    p[idx] = z;
}

// ---------------------------------------------------------------------------
// Kernel 1: atomic basis values -> _Float16 workspace, row-major (B, K)
// basisH[b*K + i*33 + d] = fup1((x[b,i]-center[d])*comp[d])
// Support |x - c_d| <= a/comp = 3/32 with center spacing 1/16 => only a
// contiguous window of <=4 centers is active per input. Each thread runs a
// fixed 4-wide window of independent Chebyshev recurrences (one v_cos_f32
// each), giving 8 independent FMAs per k-iteration for latency hiding.
// ---------------------------------------------------------------------------
__global__ void basis_kernel(const float* __restrict__ x,
                             const float* __restrict__ comp,
                             const float* __restrict__ cent,
                             _Float16* __restrict__ basisH) {
    __shared__ float sc[NSUM];     // Fourier coefficients c_k
    __shared__ float scen[DDIM];
    __shared__ float scom[DDIM];

    const int t = threadIdx.x;
    if (t < NSUM) {
        // t_k = (pi/a) * k,  c_k = sinc(t/2)^n * prod_{j=1..10} sinc(t/2^j), n=1
        const float tk = (3.14159265358979323846f / A_SUP) * (float)(t + 1);
        float c = hsinc(0.5f * tk);        // ^n with n = 1
        float p = 0.5f;
        #pragma unroll
        for (int j = 1; j <= NPROD; ++j) { c *= hsinc(tk * p); p *= 0.5f; }
        sc[t] = c;
    }
    if (t < DDIM) { scen[t] = cent[t]; scom[t] = comp[t]; }
    __syncthreads();

    const int gid = blockIdx.x * blockDim.x + t;     // gid = b*IDIM + i
    const float xv = x[gid];

    // active window of centers: d0 .. d0+3 (uniform grid)
    const float c0 = scen[0];
    const float h  = scen[1] - scen[0];
    const float w  = A_SUP / scom[0];
    int d0 = (int)ceilf((xv - w - c0) / h);
    if (d0 < 0) d0 = 0;

    float s[4], pm1[4], pm2[4], tw[4];
    bool  act[4];
    #pragma unroll
    for (int j = 0; j < 4; ++j) {
        const int d  = d0 + j;
        const bool valid = (d < DDIM);
        const int dc = valid ? d : (DDIM - 1);
        const float z = (xv - scen[dc]) * scom[dc];
        act[j] = valid && (__builtin_fabsf(z) <= A_SUP);
        // theta = (pi/a) z ; revolutions = z / (2a)
        const float c1 = __builtin_amdgcn_cosf(z * (1.0f / (2.0f * A_SUP)));
        s[j]   = fmaf(sc[0], c1, 0.5f);
        pm1[j] = c1; pm2[j] = 1.0f; tw[j] = 2.0f * c1;
    }

    #pragma unroll 2
    for (int k = 1; k < NSUM; ++k) {
        const float ck = sc[k];
        #pragma unroll
        for (int j = 0; j < 4; ++j) {
            const float pk = fmaf(tw[j], pm1[j], -pm2[j]);   // cos((k+1)theta)
            s[j] = fmaf(ck, pk, s[j]);
            pm2[j] = pm1[j]; pm1[j] = pk;
        }
    }

    _Float16* o = basisH + (size_t)gid * DDIM;
    #pragma unroll
    for (int j = 0; j < 4; ++j)
        if (act[j]) o[d0 + j] = (_Float16)(s[j] * (1.0f / A_SUP));
}

// ---------------------------------------------------------------------------
// Kernel 2: permute atomic_coeffs (I, O, D) f32 -> coefT (O, K) f16
// coefT[o*K + i*33 + d] = coeffs[(i*O + o)*D + d]
// ---------------------------------------------------------------------------
__global__ void pack_coeffs(const float* __restrict__ coeffs,
                            _Float16* __restrict__ coefT) {
    const int idx = blockIdx.x * blockDim.x + threadIdx.x;
    if (idx >= ODIM * KDIM) return;
    const int o  = idx / KDIM;
    const int kk = idx - o * KDIM;
    const int i  = kk / DDIM;
    const int d  = kk - i * DDIM;
    coefT[idx] = (_Float16)coeffs[(i * ODIM + o) * DDIM + d];
}

// ---------------------------------------------------------------------------
// Kernel 3: y[b,o] = sum_k basisH[b,k] * coefT[o,k]
// M=256, N=256, K=4224. One wave per 16x16 tile, v_wmma_f32_16x16x32_f16.
// Fragment layouts per CDNA5 ISA 7.12.2 (wave32).
// ---------------------------------------------------------------------------
__global__ void gemm_wmma(const _Float16* __restrict__ A,   // (256, K) row-major
                          const _Float16* __restrict__ Bt,  // (256, K) row-major (N-major)
                          float* __restrict__ C) {          // (256, 256)
    const int wave = (blockIdx.x * blockDim.x + threadIdx.x) >> 5;  // 0..255
    const int lane = threadIdx.x & 31;
    const int tm = wave >> 4;        // 0..15
    const int tn = wave & 15;        // 0..15
    const int l  = lane & 15;
    const int hi = lane >> 4;        // 0 or 1

    // A frag: lane (l, hi) holds row (tm*16+l), K = {hi*8..+7, hi*8+16..+23}
    const _Float16* arow = A  + (size_t)(tm * 16 + l) * KDIM + hi * 8;
    // B frag: lane (l, hi) holds col (tn*16+l), K = {hi*8..+7, hi*8+16..+23}
    const _Float16* bcol = Bt + (size_t)(tn * 16 + l) * KDIM + hi * 8;

    v8f acc = {};
    for (int k0 = 0; k0 < KDIM; k0 += 32) {
        FragH a, b;
        a.f4[0] = *(const float4*)(arow + k0);        // halves 0..7
        a.f4[1] = *(const float4*)(arow + k0 + 16);   // halves 8..15
        b.f4[0] = *(const float4*)(bcol + k0);        // halves 0..7
        b.f4[1] = *(const float4*)(bcol + k0 + 16);   // halves 8..15
        acc = __builtin_amdgcn_wmma_f32_16x16x32_f16(
            /*neg_a=*/false, a.v, /*neg_b=*/false, b.v,
            /*c_mod=*/(short)0, acc, /*reuse_a=*/false, /*reuse_b=*/false);
    }

    // C/D layout: VGPR r -> M = r (lanes 0-15) / r+8 (lanes 16-31), N = l
    float* crow = C + (size_t)(tm * 16 + hi * 8) * ODIM + tn * 16 + l;
    #pragma unroll
    for (int r = 0; r < 8; ++r) crow[(size_t)r * ODIM] = acc[r];
}

extern "C" void kernel_launch(void* const* d_in, const int* in_sizes, int n_in,
                              void* d_out, int out_size, void* d_ws, size_t ws_size,
                              hipStream_t stream) {
    const float* x      = (const float*)d_in[0];   // (256, 128)
    const float* coeffs = (const float*)d_in[1];   // (128, 256, 33)
    const float* comp   = (const float*)d_in[2];   // (33,)
    const float* cent   = (const float*)d_in[3];   // (33,)
    float* y = (float*)d_out;                      // (256, 256)

    _Float16* basisH = (_Float16*)d_ws;                                    // 256*4224 halves
    _Float16* coefT  = (_Float16*)((char*)d_ws + (size_t)BDIM * KDIM * 2); // 256*4224 halves

    // 0) zero basis workspace: 256*4224*2 B = 135168 float4s
    zero_basis<<<(BDIM * KDIM * 2 / 16) / 256, 256, 0, stream>>>((float4*)basisH);

    // 1) basis: one thread per (b,i); 128 blocks x 256 threads
    basis_kernel<<<(BDIM * IDIM) / 256, 256, 0, stream>>>(x, comp, cent, basisH);

    // 2) coeff permute+downcast: 256*4224 elements
    pack_coeffs<<<(ODIM * KDIM + 255) / 256, 256, 0, stream>>>(coeffs, coefT);

    // 3) WMMA GEMM: 256 waves = 32 blocks x 256 threads (8 waves/block)
    gemm_wmma<<<32, 256, 0, stream>>>(basisH, coefT, y);
}
